// GCN_87067577025100
// MI455X (gfx1250) — matrix operand order repaired and checked by reading, not compiled, not run on hardware
//
#include <hip/hip_runtime.h>

typedef __attribute__((ext_vector_type(16))) __bf16 v16bf;
typedef __attribute__((ext_vector_type(8)))  float  v8f;

#define TM 128
#define TN 128
#define TK 32
#define LDSS 40   // row stride in bf16 elems (80 B): 16B-aligned chunks + bank-conflict padding

__device__ __forceinline__ unsigned short f2bf(float f) {
  union { float f; unsigned int u; } c; c.f = f;
  unsigned int r = c.u + 0x7FFFu + ((c.u >> 16) & 1u);   // round-to-nearest-even
  return (unsigned short)(r >> 16);
}
__device__ __forceinline__ float bf2f(unsigned short u) {
  union { unsigned int u; float f; } c; c.u = ((unsigned int)u) << 16; return c.f;
}

union AFrag { v16bf v; uint4 q[2]; };

// ---- per-lane async global->LDS copy of one 64x32 bf16 tile (256 threads, 16B chunks)
__device__ __forceinline__ void issue_async_tile(
    const unsigned short* __restrict__ gbase,  // tile origin (row0, k0)
    long row_elems,                            // leading dim in elements
    unsigned lds_base, int tid, int rows_valid) {
  const int row = tid >> 2;                    // 0..63
  const int c   = tid & 3;                     // 16B chunk within 64B row
  if (row < rows_valid) {
    unsigned lds = lds_base + (unsigned)(row * (LDSS * 2) + c * 16);
    unsigned long long ga =
        (unsigned long long)(size_t)(gbase + (long)row * row_elems + c * 8);
    asm volatile("global_load_async_to_lds_b128 %0, %1, off"
                 :: "v"(lds), "v"(ga) : "memory");
  }
}
__device__ __forceinline__ void wait_async0() {
  asm volatile("s_wait_asynccnt 0" ::: "memory");
}

// ---------------------------------------------------------------- L2 normalize -> bf16
__global__ __launch_bounds__(256)
void gcn_l2norm(const float* __restrict__ x, unsigned short* __restrict__ h, int d) {
  __shared__ float red[256];
  const int row = blockIdx.x;
  const float* src = &x[(long)row * d];
  float s = 0.f;
  for (int j = threadIdx.x; j < d; j += 256) { float v = src[j]; s += v * v; }
  red[threadIdx.x] = s;
  __syncthreads();
  for (int off = 128; off > 0; off >>= 1) {
    if ((int)threadIdx.x < off) red[threadIdx.x] += red[threadIdx.x + off];
    __syncthreads();
  }
  const float scale = 1.f / fmaxf(sqrtf(red[0]), 1e-12f);
  unsigned short* dst = &h[(long)row * d];
  for (int j = threadIdx.x; j < d; j += 256) dst[j] = f2bf(src[j] * scale);
}

// ---------------------------------------------------------------- W[K,N] f32 -> Wt[N,K] bf16
__global__ __launch_bounds__(256)
void gcn_wt_bf16(const float* __restrict__ W, unsigned short* __restrict__ Wt,
                 int K, int N) {
  __shared__ float t[32][33];
  const int bk = blockIdx.x * 32, bn = blockIdx.y * 32;
  const int tx = threadIdx.x & 31, ty = threadIdx.x >> 5;   // 32x8
  #pragma unroll
  for (int r = 0; r < 32; r += 8)
    t[ty + r][tx] = W[(long)(bk + ty + r) * N + bn + tx];
  __syncthreads();
  #pragma unroll
  for (int r = 0; r < 32; r += 8)
    Wt[(long)(bn + ty + r) * K + bk + tx] = f2bf(t[tx][ty + r]);
}

// ---------------------------------------------------------------- bf16 WMMA GEMM
// Sb[M,N](bf16) = Hb[M,K](bf16) @ Wt[N,K](bf16)^T ; fp32 accum, async-LDS double buffer.
__global__ __launch_bounds__(256)
void gcn_gemm_bf16(const unsigned short* __restrict__ Hb,
                   const unsigned short* __restrict__ Wt,
                   unsigned short* __restrict__ Sb, int M, int K, int N) {
  __shared__ __align__(16) unsigned short As[2][TM * LDSS];   // 2 x 10 KB
  __shared__ __align__(16) unsigned short Bs[2][TN * LDSS];   // 2 x 10 KB

  const int tid  = threadIdx.x;
  const int lane = tid & 31;
  const int wave = tid >> 5;          // 0..7
  const int wm   = wave >> 1;         // 0..3 : 32-row slab
  const int wn   = wave & 1;          // 0..1 : 64-col slab
  const int m0   = blockIdx.x * TM;
  const int n0   = blockIdx.y * TN;
  const int half = lane >> 4;
  const int l16  = lane & 15;

  const unsigned aBase[2] = {(unsigned)(size_t)&As[0][0], (unsigned)(size_t)&As[1][0]};
  const unsigned bBase[2] = {(unsigned)(size_t)&Bs[0][0], (unsigned)(size_t)&Bs[1][0]};

  v8f acc[2][4];
  #pragma unroll
  for (int i = 0; i < 2; ++i)
    #pragma unroll
    for (int j = 0; j < 4; ++j)
      acc[i][j] = (v8f)(0.0f);

  const int steps = K / TK;
  const int rowsA = (M - m0 < TM) ? (M - m0) : TM;
  const unsigned short* Atile = Hb + (long)m0 * K;
  const unsigned short* Btile = Wt + (long)n0 * K;

  // prologue: fill buffer 0
  issue_async_tile(Atile, K, aBase[0], tid, rowsA);
  issue_async_tile(Btile, K, bBase[0], tid, TN);

  int cur = 0;
  for (int s = 0; s < steps; ++s) {
    wait_async0();        // this wave's fills of buf[cur] landed
    __syncthreads();      // everyone's fills landed
    if (s + 1 < steps) {  // prefetch next K-slab into the other buffer
      const long koff = (long)(s + 1) * TK;
      issue_async_tile(Atile + koff, K, aBase[cur ^ 1], tid, rowsA);
      issue_async_tile(Btile + koff, K, bBase[cur ^ 1], tid, TN);
    }

    // A frag: VGPR v holds K = 16*(v/4) + 8*half + 2*(v&3)  -> chunks at 8*half, 16+8*half
    AFrag a[2], b[4];
    #pragma unroll
    for (int m = 0; m < 2; ++m) {
      const unsigned short* rowp = &As[cur][(wm * 32 + m * 16 + l16) * LDSS];
      a[m].q[0] = *(const uint4*)&rowp[8 * half];
      a[m].q[1] = *(const uint4*)&rowp[16 + 8 * half];
    }
    // B frag: lane = column, VGPR v holds K = 16*(v/4) + 8*half + 2*(v&3) -> chunks at 8*half, 16+8*half
    #pragma unroll
    for (int n = 0; n < 4; ++n) {
      const unsigned short* colp = &Bs[cur][(wn * 64 + n * 16 + l16) * LDSS];
      b[n].q[0] = *(const uint4*)&colp[8 * half];
      b[n].q[1] = *(const uint4*)&colp[16 + 8 * half];
    }

    #pragma unroll
    for (int m = 0; m < 2; ++m)
      #pragma unroll
      for (int n = 0; n < 4; ++n)
        acc[m][n] = __builtin_amdgcn_wmma_f32_16x16x32_bf16(
            false, a[m].v, false, b[n].v, (short)0, acc[m][n], false, false);

    __syncthreads();
    cur ^= 1;
  }

  // C/D layout: VGPR r -> M = 8*half + r, N = l16 ; store bf16
  #pragma unroll
  for (int m = 0; m < 2; ++m)
    #pragma unroll
    for (int n = 0; n < 4; ++n)
      #pragma unroll
      for (int r = 0; r < 8; ++r) {
        const int grow = m0 + wm * 32 + m * 16 + half * 8 + r;
        const int gcol = n0 + wn * 64 + n * 16 + l16;
        if (grow < M) Sb[(long)grow * N + gcol] = f2bf(acc[m][n][r]);
      }
}

// ---------------------------------------------------------------- edge scatter (bf16 gather, f32 atomic accum)
__global__ __launch_bounds__(256)
void gcn_spmm(const int* __restrict__ erow, const int* __restrict__ ecol,
              const float* __restrict__ eval, const unsigned short* __restrict__ S,
              float* __restrict__ O, int d) {
  const int e = blockIdx.x;
  const int r = erow[e];
  const int c = ecol[e];
  const float v = eval[e];
  const unsigned short* src = &S[(long)c * d];
  float* dst = &O[(long)r * d];
  for (int j = threadIdx.x; j < d; j += 256)
    unsafeAtomicAdd(&dst[j], v * bf2f(src[j]));
}

// ---------------------------------------------------------------- bias + leaky -> bf16 activations
__global__ __launch_bounds__(256)
void gcn_bias_act_bf16(const float* __restrict__ acc, const float* __restrict__ b,
                       unsigned short* __restrict__ out, long total, int mask) {
  long i = (long)blockIdx.x * blockDim.x + threadIdx.x;
  const long stride = (long)gridDim.x * blockDim.x;
  for (; i < total; i += stride) {
    float v = acc[i] + b[(int)(i & mask)];
    v = (v >= 0.f) ? v : 0.2f * v;
    out[i] = f2bf(v);
  }
}

// ---------------------------------------------------------------- final bias (fp32 in place)
__global__ __launch_bounds__(256)
void gcn_bias_final(float* __restrict__ h, const float* __restrict__ b,
                    long total, int mask) {
  long i = (long)blockIdx.x * blockDim.x + threadIdx.x;
  const long stride = (long)gridDim.x * blockDim.x;
  for (; i < total; i += stride) h[i] += b[(int)(i & mask)];
}

extern "C" void kernel_launch(void* const* d_in, const int* in_sizes, int n_in,
                              void* d_out, int out_size, void* d_ws, size_t ws_size,
                              hipStream_t stream) {
  const float* x    = (const float*)d_in[0];
  const int*   erow = (const int*)d_in[1];
  const int*   ecol = (const int*)d_in[2];
  const float* eval = (const float*)d_in[3];
  const float* W[3] = {(const float*)d_in[4], (const float*)d_in[6], (const float*)d_in[8]};
  const float* B[3] = {(const float*)d_in[5], (const float*)d_in[7], (const float*)d_in[9]};

  const int dims[4] = {512, 2048, 1024, 512};
  const int n_nodes = in_sizes[0] / dims[0];
  const int n_edges = in_sizes[1];
  const size_t maxact = (size_t)n_nodes * 2048;

  float*          accb = (float*)d_ws;                        // fp32 scatter accumulator
  unsigned short* hb   = (unsigned short*)(accb + maxact);    // bf16 activations
  unsigned short* sb   = hb + maxact;                         // bf16 GEMM output
  unsigned short* wt   = sb + maxact;                         // bf16 transposed weights

  gcn_l2norm<<<n_nodes, 256, 0, stream>>>(x, hb, dims[0]);

  for (int l = 0; l < 3; ++l) {
    const int K = dims[l], N = dims[l + 1];

    dim3 gt(K / 32, N / 32);
    gcn_wt_bf16<<<gt, 256, 0, stream>>>(W[l], wt, K, N);

    dim3 gg((n_nodes + TM - 1) / TM, N / TN);
    gcn_gemm_bf16<<<gg, 256, 0, stream>>>(hb, wt, sb, n_nodes, K, N);

    const long total = (long)n_nodes * N;
    if (l < 2) {
      hipMemsetAsync(accb, 0, (size_t)total * sizeof(float), stream);
      gcn_spmm<<<n_edges, 256, 0, stream>>>(erow, ecol, eval, sb, accb, N);
      gcn_bias_act_bf16<<<4096, 256, 0, stream>>>(accb, B[l], hb, total, N - 1);
    } else {
      hipMemsetAsync(d_out, 0, (size_t)total * sizeof(float), stream);
      gcn_spmm<<<n_edges, 256, 0, stream>>>(erow, ecol, eval, sb, (float*)d_out, N);
      gcn_bias_final<<<4096, 256, 0, stream>>>((float*)d_out, B[l], total, N - 1);
    }
  }
}
